// RNADesignNetwork_38096359915723
// MI455X (gfx1250) — matrix operand order repaired and checked by reading, not compiled, not run on hardware
//
#include <hip/hip_runtime.h>
#include <hip/hip_bf16.h>

typedef _Float16 f16_t;
typedef _Float16 half16 __attribute__((ext_vector_type(16)));
typedef _Float16 half8  __attribute__((ext_vector_type(8)));
typedef _Float16 half4  __attribute__((ext_vector_type(4)));
typedef float    floatx8 __attribute__((ext_vector_type(8)));

#define BN_EPS_F 1e-5f

__device__ __forceinline__ floatx8 wmma_f16(half16 a, half16 b, floatx8 c) {
  // D = A(16x32 f16) x B(32x16 f16) + C(16x16 f32)
  return __builtin_amdgcn_wmma_f32_16x16x32_f16(false, a, false, b, (short)0, c,
                                                false, false);
}

__device__ __forceinline__ half16 cat16(half8 lo, half8 hi) {
  return __builtin_shufflevector(lo, hi, 0, 1, 2, 3, 4, 5, 6, 7, 8, 9, 10, 11,
                                 12, 13, 14, 15);
}

// ---------------------------------------------------------------- zero
__global__ void k_zero(float* p, long n) {
  long i = (long)blockIdx.x * blockDim.x + threadIdx.x;
  long stride = (long)gridDim.x * blockDim.x;
  for (; i < n; i += stride) p[i] = 0.f;
}

// ----------------------------------------------------- BN fold (eval mode)
// scale = g * rsqrt(v+eps); bias = (b1 - m) * scale + bt
__global__ void k_bnprep(const float* g0, const float* bt0, const float* m0,
                         const float* v0, const float* b10, const float* g1,
                         const float* bt1, const float* m1, const float* v1,
                         const float* b11, float* scale0, float* bias0,
                         float* scale1, float* bias1) {
  int t = threadIdx.x;
  if (t < 128) {
    float s = g0[t] * rsqrtf(v0[t] + BN_EPS_F);
    scale0[t] = s;
    bias0[t] = (b10[t] - m0[t]) * s + bt0[t];
  } else if (t < 192) {
    int n = t - 128;
    float s = g1[n] * rsqrtf(v1[n] + BN_EPS_F);
    scale1[n] = s;
    bias1[n] = (b11[n] - m1[n]) * s + bt1[n];
  }
}

// -------------------------------------------- pack W into WMMA B-fragments
// B layout (16-bit, 32x16): e<8 -> k_local=(lane>>4)*8+e, e>=8 -> 16+(lane>>4)*8+(e-8);
// n = nt*16 + (lane&15).
// dst[frag][lane][e], frag = kt*ntiles+nt. Optional per-column scale (BN fold).
__global__ void k_pack(const float* W, int Ksrc, int Ncols, int ktiles,
                       const float* scale, f16_t* dst) {
  int ntiles = Ncols >> 4;
  int total = ktiles * ntiles * 512;
  for (int t = blockIdx.x * blockDim.x + threadIdx.x; t < total;
       t += gridDim.x * blockDim.x) {
    int e = t & 15;
    int lane = (t >> 4) & 31;
    int frag = t >> 9;
    int kt = frag / ntiles, nt = frag % ntiles;
    int n = nt * 16 + (lane & 15);
    int k = kt * 32 + ((lane >> 4) << 3) + (e & 7) + ((e >> 3) << 4);
    float v = 0.f;
    if (k < Ksrc) v = W[(long)k * Ncols + n] * (scale ? scale[n] : 1.f);
    dst[t] = (f16_t)v;
  }
}

// ------------------------------------------- layer-0 edge message + scatter
// msg = relu(x[src] + ea@We0 + be0) ; atomic add into aggr0[dst][9]
__global__ void k_scatter0(const float* nf, const int* ei, const float* ea,
                           const float* We, const float* be, float* aggr,
                           int E_, int N_) {
  __shared__ float sW[36];
  __shared__ float sB[9];
  int t = threadIdx.x;
  if (t < 36) sW[t] = We[t];
  if (t < 9) sB[t] = be[t];
  __syncthreads();
  int e = blockIdx.x * blockDim.x + t;
  if (e >= E_) return;
  int src = ei[e], dst = ei[E_ + e];
  const float4 av = *(const float4*)(ea + (long)e * 4);
  const float* xs = nf + (long)src * 9;
  float* ad = aggr + (long)dst * 9;
#pragma unroll
  for (int c = 0; c < 9; ++c) {
    float m = xs[c] + av.x * sW[0 * 9 + c] + av.y * sW[1 * 9 + c] +
              av.z * sW[2 * 9 + c] + av.w * sW[3 * 9 + c] + sB[c];
    atomicAdd(&ad[c], fmaxf(m, 0.f));
  }
}

// ------------------------------------------- layer-1 edge message + scatter
// one wave per edge, 4 channels per lane; x0 gathered as f16 (halved traffic)
__global__ void k_scatter1(const f16_t* x0h, const int* ei, const float* ea,
                           const float* We, const float* be, float* aggr,
                           int E_) {
  int t = blockIdx.x * blockDim.x + threadIdx.x;
  int e = t >> 5;
  if (e >= E_) return;
  int lane = t & 31;
  int src = ei[e], dst = ei[E_ + e];
  const float4 av = *(const float4*)(ea + (long)e * 4);
  int c0 = lane * 4;
  half4 xv = *(const half4*)(x0h + (long)src * 128 + c0);
  float* ad = aggr + (long)dst * 128 + c0;
#pragma unroll
  for (int j = 0; j < 4; ++j) {
    int c = c0 + j;
    float m = (float)xv[j] + av.x * We[c] + av.y * We[128 + c] +
              av.z * We[256 + c] + av.w * We[384 + c] + be[c];
    atomicAdd(&ad[j], fmaxf(m, 0.f));
  }
}

// --------------------------------------- layer-0 node MLP: 9 -> 128 -> 128
// one 16-node tile per wave; GEMM1: 1 ktile x 8 ntiles, GEMM2: 4 x 8 WMMA
__global__ void __launch_bounds__(128) k_mlp0(const float* nf,
                                              const float* aggr0,
                                              const f16_t* w1p,
                                              const float* bias0,
                                              const f16_t* w2p, const float* b2,
                                              f16_t* x0h, int N_) {
  __shared__ f16_t lsA[4][16 * 32];
  __shared__ f16_t lh[4][16 * 128];
  const int lane = threadIdx.x & 31;
  const int wave = threadIdx.x >> 5;
  const int base = blockIdx.x * 64 + wave * 16;
  const int half_ = lane >> 4;
  const int nl = lane & 15;
  const floatx8 vzero = {0.f, 0.f, 0.f, 0.f, 0.f, 0.f, 0.f, 0.f};

  // stage A1 = x + aggr0 (K padded 9 -> 32, zeros)
  for (int idx = lane; idx < 512; idx += 32) {
    int m = idx >> 5, k = idx & 31;
    int node = base + m;
    float v = 0.f;
    if (k < 9 && node < N_)
      v = nf[(long)node * 9 + k] + aggr0[(long)node * 9 + k];
    lsA[wave][idx] = (f16_t)v;
  }
  __syncthreads();

  // A fragment: lane -> row m=nl; e<8 -> k=half*8+e ; e>=8 -> k=half*8+8+e
  half8 lo = *(const half8*)&lsA[wave][nl * 32 + half_ * 8];
  half8 hi = *(const half8*)&lsA[wave][nl * 32 + half_ * 8 + 16];
  half16 a1 = cat16(lo, hi);

  floatx8 acc[8];
#pragma unroll
  for (int nt = 0; nt < 8; ++nt) {
    half16 b = *(const half16*)(w1p + nt * 512 + lane * 16);
    acc[nt] = wmma_f16(a1, b, vzero);
  }
  // fused BN bias + relu, stage h[16][128] as f16
#pragma unroll
  for (int nt = 0; nt < 8; ++nt) {
    int col = nt * 16 + nl;
    float bb = bias0[col];
#pragma unroll
    for (int r = 0; r < 8; ++r)
      lh[wave][(r + 8 * half_) * 128 + col] = (f16_t)fmaxf(acc[nt][r] + bb, 0.f);
  }
  __syncthreads();

  half16 a2[4];
#pragma unroll
  for (int kt = 0; kt < 4; ++kt) {
    half8 l2 = *(const half8*)&lh[wave][nl * 128 + kt * 32 + half_ * 8];
    half8 h2 = *(const half8*)&lh[wave][nl * 128 + kt * 32 + half_ * 8 + 16];
    a2[kt] = cat16(l2, h2);
  }
  floatx8 acc2[8];
#pragma unroll
  for (int nt = 0; nt < 8; ++nt) acc2[nt] = vzero;
#pragma unroll
  for (int kt = 0; kt < 4; ++kt) {
#pragma unroll
    for (int nt = 0; nt < 8; ++nt) {
      half16 b = *(const half16*)(w2p + (kt * 8 + nt) * 512 + lane * 16);
      acc2[nt] = wmma_f16(a2[kt], b, acc2[nt]);
    }
  }
#pragma unroll
  for (int nt = 0; nt < 8; ++nt) {
    int col = nt * 16 + nl;
    float bb = b2[col];
#pragma unroll
    for (int r = 0; r < 8; ++r) {
      int node = base + r + 8 * half_;
      if (node < N_)
        x0h[(long)node * 128 + col] = (f16_t)fmaxf(acc2[nt][r] + bb, 0.f);
    }
  }
}

// ---------------- layer-1 node MLP (128->64->64) + fused location head
__global__ void __launch_bounds__(128) k_mlp1(
    const f16_t* x0h, const float* aggr1, const f16_t* w1p, const float* bias1,
    const f16_t* w2p, const float* b2, const f16_t* lap, const float* la_b1,
    const float* la_w2, const float* la_b2, float* outX, float* outLoc,
    int N_) {
  __shared__ f16_t lhA[4][16 * 128];
  __shared__ f16_t lhB[4][16 * 64];
  __shared__ float lsf[4][16 * 16];
  const int lane = threadIdx.x & 31;
  const int wave = threadIdx.x >> 5;
  const int base = blockIdx.x * 64 + wave * 16;
  const int half_ = lane >> 4;
  const int nl = lane & 15;
  const floatx8 vzero = {0.f, 0.f, 0.f, 0.f, 0.f, 0.f, 0.f, 0.f};

  // stage A = x0 + aggr1, f16 [16][128]
  for (int idx = lane; idx < 2048; idx += 32) {
    int m = idx >> 7, c = idx & 127;
    int node = base + m;
    float v = 0.f;
    if (node < N_)
      v = (float)x0h[(long)node * 128 + c] + aggr1[(long)node * 128 + c];
    lhA[wave][idx] = (f16_t)v;
  }
  __syncthreads();

  half16 a[4];
#pragma unroll
  for (int kt = 0; kt < 4; ++kt) {
    half8 l2 = *(const half8*)&lhA[wave][nl * 128 + kt * 32 + half_ * 8];
    half8 h2 = *(const half8*)&lhA[wave][nl * 128 + kt * 32 + half_ * 8 + 16];
    a[kt] = cat16(l2, h2);
  }
  floatx8 acc[4];
#pragma unroll
  for (int nt = 0; nt < 4; ++nt) acc[nt] = vzero;
#pragma unroll
  for (int kt = 0; kt < 4; ++kt) {
#pragma unroll
    for (int nt = 0; nt < 4; ++nt) {
      half16 b = *(const half16*)(w1p + (kt * 4 + nt) * 512 + lane * 16);
      acc[nt] = wmma_f16(a[kt], b, acc[nt]);
    }
  }
#pragma unroll
  for (int nt = 0; nt < 4; ++nt) {
    int col = nt * 16 + nl;
    float bb = bias1[col];
#pragma unroll
    for (int r = 0; r < 8; ++r)
      lhB[wave][(r + 8 * half_) * 64 + col] = (f16_t)fmaxf(acc[nt][r] + bb, 0.f);
  }
  __syncthreads();

  half16 a3[2];
#pragma unroll
  for (int kt = 0; kt < 2; ++kt) {
    half8 l2 = *(const half8*)&lhB[wave][nl * 64 + kt * 32 + half_ * 8];
    half8 h2 = *(const half8*)&lhB[wave][nl * 64 + kt * 32 + half_ * 8 + 16];
    a3[kt] = cat16(l2, h2);
  }
  floatx8 acc3[4];
#pragma unroll
  for (int nt = 0; nt < 4; ++nt) acc3[nt] = vzero;
#pragma unroll
  for (int kt = 0; kt < 2; ++kt) {
#pragma unroll
    for (int nt = 0; nt < 4; ++nt) {
      half16 b = *(const half16*)(w2p + (kt * 4 + nt) * 512 + lane * 16);
      acc3[nt] = wmma_f16(a3[kt], b, acc3[nt]);
    }
  }
  // write x1 (fp32 output) + restage f16 for the location head (reuse lhA)
  f16_t* lhX = lhA[wave];
#pragma unroll
  for (int nt = 0; nt < 4; ++nt) {
    int col = nt * 16 + nl;
    float bb = b2[col];
#pragma unroll
    for (int r = 0; r < 8; ++r) {
      int row = r + 8 * half_;
      int node = base + row;
      float v = fmaxf(acc3[nt][r] + bb, 0.f);
      lhX[row * 64 + col] = (f16_t)v;
      if (node < N_) outX[(long)node * 64 + col] = v;
    }
  }
  __syncthreads();

  // location head: relu(x1 @ la_W1 + la_b1) @ la_W2 + la_b2
  half16 a4[2];
#pragma unroll
  for (int kt = 0; kt < 2; ++kt) {
    half8 l2 = *(const half8*)&lhX[nl * 64 + kt * 32 + half_ * 8];
    half8 h2 = *(const half8*)&lhX[nl * 64 + kt * 32 + half_ * 8 + 16];
    a4[kt] = cat16(l2, h2);
  }
  floatx8 accH = vzero;
#pragma unroll
  for (int kt = 0; kt < 2; ++kt) {
    half16 b = *(const half16*)(lap + kt * 512 + lane * 16);
    accH = wmma_f16(a4[kt], b, accH);
  }
  float w2v = la_w2[nl];
  float bb = la_b1[nl];
#pragma unroll
  for (int r = 0; r < 8; ++r)
    lsf[wave][(r + 8 * half_) * 16 + nl] = fmaxf(accH[r] + bb, 0.f) * w2v;
  __syncthreads();
  if (lane < 16) {
    int node = base + lane;
    if (node < N_) {
      float s = la_b2[0];
#pragma unroll
      for (int n = 0; n < 16; ++n) s += lsf[wave][lane * 16 + n];
      outLoc[node] = s;
    }
  }
}

// ------------------------------------------- mutation head (B=64, tiny)
__global__ void k_mut(const float* outX, const int* tgt, const float* W1,
                      const float* b1, const float* W2, const float* b2,
                      float* outM, int B_) {
  int b = blockIdx.x * blockDim.x + threadIdx.x;
  if (b >= B_) return;
  int node = tgt[b];
  const float* sel = outX + (long)node * 64;
  float h[16];
#pragma unroll
  for (int k = 0; k < 16; ++k) {
    float s = b1[k];
    for (int j = 0; j < 64; ++j) s += sel[j] * W1[j * 16 + k];
    h[k] = fmaxf(s, 0.f);
  }
#pragma unroll
  for (int c = 0; c < 4; ++c) {
    float s = b2[c];
#pragma unroll
    for (int k = 0; k < 16; ++k) s += h[k] * W2[k * 4 + c];
    outM[b * 4 + c] = s;
  }
}

extern "C" void kernel_launch(void* const* d_in, const int* in_sizes, int n_in,
                              void* d_out, int out_size, void* d_ws,
                              size_t ws_size, hipStream_t stream) {
  const float* nf = (const float*)d_in[0];
  const int* ei = (const int*)d_in[1];
  const float* ea = (const float*)d_in[2];
  const int* tgt = (const int*)d_in[5];
  const float* l0_We = (const float*)d_in[6];
  const float* l0_be = (const float*)d_in[7];
  const float* l0_W1 = (const float*)d_in[8];
  const float* l0_b1 = (const float*)d_in[9];
  const float* l0_g = (const float*)d_in[10];
  const float* l0_bt = (const float*)d_in[11];
  const float* l0_m = (const float*)d_in[12];
  const float* l0_v = (const float*)d_in[13];
  const float* l0_W2 = (const float*)d_in[14];
  const float* l0_b2 = (const float*)d_in[15];
  const float* l1_We = (const float*)d_in[16];
  const float* l1_be = (const float*)d_in[17];
  const float* l1_W1 = (const float*)d_in[18];
  const float* l1_b1 = (const float*)d_in[19];
  const float* l1_g = (const float*)d_in[20];
  const float* l1_bt = (const float*)d_in[21];
  const float* l1_m = (const float*)d_in[22];
  const float* l1_v = (const float*)d_in[23];
  const float* l1_W2 = (const float*)d_in[24];
  const float* l1_b2 = (const float*)d_in[25];
  const float* la_W1 = (const float*)d_in[26];
  const float* la_b1 = (const float*)d_in[27];
  const float* la_W2 = (const float*)d_in[28];
  const float* la_b2 = (const float*)d_in[29];
  const float* ma_W1 = (const float*)d_in[30];
  const float* ma_b1 = (const float*)d_in[31];
  const float* ma_W2 = (const float*)d_in[32];
  const float* ma_b2 = (const float*)d_in[33];

  const int N_ = in_sizes[0] / 9;
  const int E_ = in_sizes[2] / 4;
  const int B_ = in_sizes[5];

  // workspace carve-out (256B aligned)
  char* ws = (char*)d_ws;
  size_t off = 0;
  auto carve = [&](size_t bytes) -> void* {
    void* p = ws + off;
    off = (off + bytes + 255) & ~(size_t)255;
    return p;
  };
  float* aggr0 = (float*)carve((size_t)N_ * 9 * sizeof(float));
  float* aggr1 = (float*)carve((size_t)N_ * 128 * sizeof(float));
  f16_t* x0h = (f16_t*)carve((size_t)N_ * 128 * sizeof(f16_t));
  float* scale0 = (float*)carve(128 * sizeof(float));
  float* bias0 = (float*)carve(128 * sizeof(float));
  float* scale1 = (float*)carve(64 * sizeof(float));
  float* bias1 = (float*)carve(64 * sizeof(float));
  f16_t* w1p0 = (f16_t*)carve(8 * 512 * sizeof(f16_t));    // 9(->32) x 128
  f16_t* w2p0 = (f16_t*)carve(32 * 512 * sizeof(f16_t));   // 128 x 128
  f16_t* w1p1 = (f16_t*)carve(16 * 512 * sizeof(f16_t));   // 128 x 64
  f16_t* w2p1 = (f16_t*)carve(8 * 512 * sizeof(f16_t));    // 64 x 64
  f16_t* lap = (f16_t*)carve(2 * 512 * sizeof(f16_t));     // 64 x 16

  float* outLoc = (float*)d_out;            // [N]
  float* outM = outLoc + N_;                // [B*4]
  float* outX = outM + (size_t)B_ * 4;      // [N*64]

  // 1) zero aggregation buffers
  k_zero<<<2048, 256, 0, stream>>>(aggr0, (long)N_ * 9);
  k_zero<<<4096, 256, 0, stream>>>(aggr1, (long)N_ * 128);
  // 2) fold BN, pack weights into WMMA B-fragment layout (f16)
  k_bnprep<<<1, 192, 0, stream>>>(l0_g, l0_bt, l0_m, l0_v, l0_b1, l1_g, l1_bt,
                                  l1_m, l1_v, l1_b1, scale0, bias0, scale1,
                                  bias1);
  k_pack<<<16, 256, 0, stream>>>(l0_W1, 9, 128, 1, scale0, w1p0);
  k_pack<<<64, 256, 0, stream>>>(l0_W2, 128, 128, 4, (const float*)nullptr,
                                 w2p0);
  k_pack<<<32, 256, 0, stream>>>(l1_W1, 128, 64, 4, scale1, w1p1);
  k_pack<<<16, 256, 0, stream>>>(l1_W2, 64, 64, 2, (const float*)nullptr,
                                 w2p1);
  k_pack<<<4, 256, 0, stream>>>(la_W1, 64, 16, 2, (const float*)nullptr, lap);
  // 3) layer 0
  k_scatter0<<<(E_ + 255) / 256, 256, 0, stream>>>(nf, ei, ea, l0_We, l0_be,
                                                   aggr0, E_, N_);
  k_mlp0<<<(N_ + 63) / 64, 128, 0, stream>>>(nf, aggr0, w1p0, bias0, w2p0,
                                             l0_b2, x0h, N_);
  // 4) layer 1
  k_scatter1<<<(E_ + 7) / 8, 256, 0, stream>>>(x0h, ei, ea, l1_We, l1_be,
                                               aggr1, E_);
  k_mlp1<<<(N_ + 63) / 64, 128, 0, stream>>>(x0h, aggr1, w1p1, bias1, w2p1,
                                             l1_b2, lap, la_b1, la_W2, la_b2,
                                             outX, outLoc, N_);
  // 5) mutation head
  k_mut<<<(B_ + 63) / 64, 64, 0, stream>>>(outX, tgt, ma_W1, ma_b1, ma_W2,
                                           ma_b2, outM, B_);
}
